// SSMBlock_38371237822921
// MI455X (gfx1250) — matrix operand order repaired and checked by reading, not compiled, not run on hardware
//
#include <hip/hip_runtime.h>

// ---------------- problem constants (match reference setup_inputs) ----------
#define B_    2
#define L_    4096
#define D_    1024
#define DIN   2048
#define DST   128
#define NH    32
#define HD    64
#define CONVD 2304        // DIN + 2*DST
#define DPROJ 4384        // 2*DIN + 2*DST + NH
#define NQ    64          // chunk length
#define NC    64          // chunks per sequence (L/NQ)
#define KCONV 4

// ---------------- WMMA types & helpers --------------------------------------
typedef __attribute__((ext_vector_type(16))) __bf16 v16bf;
typedef __attribute__((ext_vector_type(8)))  __bf16 bf16x8;
typedef __attribute__((ext_vector_type(8)))  float  v8f;

#define WMMA_BF16(a,b,c) __builtin_amdgcn_wmma_f32_16x16x32_bf16(false,(a),false,(b),(short)0,(c),false,false)

__device__ __forceinline__ unsigned short f2bf(float f){
  unsigned x = __builtin_bit_cast(unsigned, f);
  x += 0x7fffu + ((x >> 16) & 1u);               // round-to-nearest-even
  return (unsigned short)(x >> 16);
}
__device__ __forceinline__ float bf2f(unsigned short h){
  unsigned x = ((unsigned)h) << 16;
  return __builtin_bit_cast(float, x);
}

// A fragment: 16x32 bf16, source row-major [M x K] with K contiguous.
// lane l: row = l%16 ; halves: k in [8h,8h+8) and [16+8h,16+8h+8)   (ISA 7.12.2)
__device__ __forceinline__ v16bf frag_a(const unsigned short* base, int ld, int lane){
  const int r = lane & 15, h = lane >> 4;
  const __bf16* p = (const __bf16*)base + (long)r * ld + 8 * h;
  bf16x8 lo = *(const bf16x8*)(p);
  bf16x8 hi = *(const bf16x8*)(p + 16);
  v16bf f;
#pragma unroll
  for (int i = 0; i < 8; i++){ f[i] = lo[i]; f[8+i] = hi[i]; }
  return f;
}
// B fragment: 32x16 bf16 where B[k,n] is supplied as row-major [N x K] rows.
// lane l: col n = l%16 ; halves: k in [8h,8h+8) and [16+8h,16+8h+8)
__device__ __forceinline__ v16bf frag_b(const unsigned short* base, int ld, int lane){
  const int r = lane & 15, h = lane >> 4;
  const __bf16* p = (const __bf16*)base + (long)r * ld + 8 * h;
  bf16x8 lo = *(const bf16x8*)(p);
  bf16x8 hi = *(const bf16x8*)(p + 16);
  v16bf f;
#pragma unroll
  for (int i = 0; i < 8; i++){ f[i] = lo[i]; f[8+i] = hi[i]; }
  return f;
}

// ---- CDNA5 async global->LDS copy (ASYNCcnt-tracked, no VGPR data path) ----
// lds_off: wave-relative LDS byte address = low 32 bits of generic LDS pointer
__device__ __forceinline__ void async_copy_b128(unsigned lds_off, const void* gptr){
  asm volatile("global_load_async_to_lds_b128 %0, %1, off"
               :: "v"(lds_off), "v"(gptr) : "memory");
}
__device__ __forceinline__ void wait_async_le5(){
  asm volatile("s_wait_asynccnt 0x5" ::: "memory");
}
__device__ __forceinline__ void wait_async_0(){
  asm volatile("s_wait_asynccnt 0x0" ::: "memory");
}
__device__ __forceinline__ unsigned lds_addr32(const void* p){
  return (unsigned)(unsigned long long)p;   // LDS generic addr: low 32 bits = offset
}

__device__ __forceinline__ float block_reduce(float v, float* sbuf){
  __syncthreads();
  const int lane = threadIdx.x & 31, w = threadIdx.x >> 5;
#pragma unroll
  for (int off = 16; off > 0; off >>= 1) v += __shfl_down(v, off, 32);
  if (lane == 0) sbuf[w] = v;
  __syncthreads();
  if (w == 0){
    const int nw = blockDim.x >> 5;
    float r = (lane < nw) ? sbuf[lane] : 0.f;
#pragma unroll
    for (int off = 16; off > 0; off >>= 1) r += __shfl_down(r, off, 32);
    if (lane == 0) sbuf[0] = r;
  }
  __syncthreads();
  return sbuf[0];
}

// ---------------- kernels ---------------------------------------------------

__global__ void cvt_bf16_kernel(const float* __restrict__ src,
                                unsigned short* __restrict__ dst, long n){
  long i = (long)blockIdx.x * blockDim.x + threadIdx.x;
  if (i < n) dst[i] = f2bf(src[i]);
}

// C[M,N](f32) = A[M,K](bf16) * B[N,K](bf16)^T
// Block tile 128x32, 4 waves (each one 32x32 WMMA tile), K-step 32.
// Operand panels staged to LDS by global_load_async_to_lds_b128, double-buffered:
// while stage ks+1 is in flight, WMMAs consume stage ks; s_wait_asynccnt 5
// (5 async issues/stage/wave, in-order completion) gates buffer reuse.
__global__ void __launch_bounds__(128)
gemm_bf16_wmma(const unsigned short* __restrict__ A,
               const unsigned short* __restrict__ Bm,
               float* __restrict__ C,
               int M, int N, int K, int lda, int ldb, int ldc){
  __shared__ __align__(16) unsigned short Abuf[2][128][32];
  __shared__ __align__(16) unsigned short Bbuf[2][32][32];
  const int tid = threadIdx.x, lane = tid & 31, wave = tid >> 5;
  const int nb = N >> 5;                    // N / 32
  const int m0 = (blockIdx.x / nb) * 128;
  const int n0 = (blockIdx.x % nb) * 32;

  const int brow = tid & 31, bchunk = tid >> 5;          // B: 32 rows x 4 chunks
  const unsigned short* gA = A  + (long)(m0 + tid)  * lda;   // one A row / thread
  const unsigned short* gB = Bm + (long)(n0 + brow) * ldb + bchunk * 8;

  const int NK = K >> 5;
  v8f acc00 = {}, acc01 = {}, acc10 = {}, acc11 = {};

  // stage 0
  {
#pragma unroll
    for (int c = 0; c < 4; c++)
      async_copy_b128(lds_addr32(&Abuf[0][tid][c * 8]), (const void*)(gA + c * 8));
    async_copy_b128(lds_addr32(&Bbuf[0][brow][bchunk * 8]), (const void*)gB);
  }
  for (int ks = 0; ks < NK; ks++){
    const int cur = ks & 1;
    if (ks + 1 < NK){
      const int k1 = (ks + 1) << 5;
#pragma unroll
      for (int c = 0; c < 4; c++)
        async_copy_b128(lds_addr32(&Abuf[cur ^ 1][tid][c * 8]),
                        (const void*)(gA + k1 + c * 8));
      async_copy_b128(lds_addr32(&Bbuf[cur ^ 1][brow][bchunk * 8]),
                      (const void*)(gB + k1));
      wait_async_le5();       // previous stage fully landed
    } else {
      wait_async_0();
    }
    __syncthreads();
    v16bf a0 = frag_a(&Abuf[cur][wave * 32][0],      32, lane);
    v16bf a1 = frag_a(&Abuf[cur][wave * 32 + 16][0], 32, lane);
    v16bf b0 = frag_b(&Bbuf[cur][0][0],  32, lane);
    v16bf b1 = frag_b(&Bbuf[cur][16][0], 32, lane);
    acc00 = WMMA_BF16(a0, b0, acc00);
    acc01 = WMMA_BF16(a0, b1, acc01);
    acc10 = WMMA_BF16(a1, b0, acc10);
    acc11 = WMMA_BF16(a1, b1, acc11);
    __syncthreads();          // readers done before next stage overwrites buf
  }
  const int half = lane >> 4, nl = lane & 15;
  const int mw = m0 + wave * 32;
#pragma unroll
  for (int r = 0; r < 8; r++){
    const int mA = mw + 8 * half + r, mB = mA + 16;
    C[(long)mA * ldc + n0 + nl]      = acc00[r];
    C[(long)mA * ldc + n0 + 16 + nl] = acc01[r];
    C[(long)mB * ldc + n0 + nl]      = acc10[r];
    C[(long)mB * ldc + n0 + 16 + nl] = acc11[r];
  }
}

// depthwise causal conv(K=4) + bias + SiLU over xBC slice of zxbcdt -> bf16
__global__ void conv_silu_kernel(const float* __restrict__ zx,
                                 const float* __restrict__ cw,
                                 const float* __restrict__ cb,
                                 unsigned short* __restrict__ xbc){
  long idx = (long)blockIdx.x * blockDim.x + threadIdx.x;
  if (idx >= (long)B_ * L_ * CONVD) return;
  const int  ch  = (int)(idx % CONVD);
  const long row = idx / CONVD;          // b*L + l
  const int  l   = (int)(row % L_);
  float acc = cb[ch];
#pragma unroll
  for (int k = 0; k < KCONV; k++){
    const int ls = l + k - (KCONV - 1);
    if (ls >= 0)
      acc += zx[(row + (long)(ls - l)) * DPROJ + DIN + ch] * cw[ch * KCONV + k];
  }
  const float s = acc / (1.f + expf(-acc));
  xbc[idx] = f2bf(s);
}

// dt = DT_MIN + (DT_MAX-DT_MIN)*sigmoid ; per-chunk inclusive cumsum of dt*A
__global__ void dt_acum_kernel(const float* __restrict__ zx,
                               const float* __restrict__ dt_bias,
                               const float* __restrict__ A_log,
                               float* __restrict__ dtb,
                               float* __restrict__ acum){
  const int blk = blockIdx.x;
  const int cc = blk % NC;
  const int h  = (blk / NC) % NH;
  const int bb = blk / (NC * NH);
  const int q  = threadIdx.x;
  const long row = (long)bb * L_ + cc * NQ + q;
  const float dtr = zx[row * DPROJ + (DIN + CONVD) + h];
  const float sg  = 1.f / (1.f + expf(-(dtr + dt_bias[h])));
  const float dtv = 0.001f + 0.099f * sg;
  dtb[row * NH + h] = dtv;
  __shared__ float s[NQ];
  s[q] = dtv * (-expf(A_log[h]));
  __syncthreads();
  for (int off = 1; off < NQ; off <<= 1){
    float v = (q >= off) ? s[q - off] : 0.f;
    __syncthreads();
    s[q] += v;
    __syncthreads();
  }
  acum[(((long)bb * NH + h) * NC + cc) * NQ + q] = s[q];
}

// per (b,chunk,head): G=C*B^T (WMMA), E=mask*exp, Yd=E*Xdt (WMMA),
// states = (decay*Xdt)^T * B (WMMA)
__global__ void __launch_bounds__(128)
chunk_fwd_kernel(const unsigned short* __restrict__ xbc,
                 const float* __restrict__ dtb,
                 const float* __restrict__ acum,
                 float* __restrict__ Ybuf,
                 float* __restrict__ states){
  const int h  = blockIdx.x & (NH - 1);
  const int cc = (blockIdx.x >> 5) & (NC - 1);
  const int bb = blockIdx.x >> 11;
  const int tid = threadIdx.x, lane = tid & 31, wave = tid >> 5;

  const unsigned short* xrow = xbc + ((long)bb * L_ + (long)cc * NQ) * CONVD;
  const unsigned short* Bc   = xrow + DIN;        // [q][DST], ld=CONVD
  const unsigned short* Ccm  = xrow + DIN + DST;  // [q][DST]
  const unsigned short* Xs   = xrow + h * HD;     // [q][HD]

  __shared__ float Ac[NQ], dtq[NQ];
  __shared__ __align__(16) unsigned short Esh[NQ][72];
  __shared__ __align__(16) unsigned short XdtT[HD][72];
  __shared__ __align__(16) unsigned short XdecT[HD][72];
  __shared__ __align__(16) unsigned short BcT[DST][72];

  if (tid < NQ){
    Ac[tid]  = acum[(((long)bb * NH + h) * NC + cc) * NQ + tid];
    dtq[tid] = dtb[((long)bb * L_ + cc * NQ + tid) * NH + h];
  }
  __syncthreads();
  const float Alast = Ac[NQ - 1];
  for (int i = tid; i < NQ * HD; i += 128){
    const int q = i >> 6, p = i & 63;
    const float xdt = bf2f(Xs[q * CONVD + p]) * dtq[q];
    XdtT[p][q]  = f2bf(xdt);
    XdecT[p][q] = f2bf(xdt * expf(Alast - Ac[q]));
  }
  for (int i = tid; i < NQ * DST; i += 128){
    const int q = i >> 7, n = i & 127;
    BcT[n][q] = Bc[q * CONVD + n];
  }
  __syncthreads();

  const int m0 = wave * 16;
  const int half = lane >> 4, nl = lane & 15;

  // ---- G = Cc * Bc^T  (64x64, K=DST) ----
  v8f g[4] = {};
  for (int k0 = 0; k0 < DST; k0 += 32){
    v16bf a = frag_a(Ccm + (long)m0 * CONVD + k0, CONVD, lane);
#pragma unroll
    for (int st = 0; st < 4; st++){
      v16bf b = frag_b(Bc + (long)(st * 16) * CONVD + k0, CONVD, lane);
      g[st] = WMMA_BF16(a, b, g[st]);
    }
  }
  // ---- E = tril-mask * exp(Ac[q]-Ac[s]) * G -> LDS bf16 ----
#pragma unroll
  for (int st = 0; st < 4; st++){
#pragma unroll
    for (int r = 0; r < 8; r++){
      const int q = m0 + 8 * half + r;
      const int s = st * 16 + nl;
      const float v = (s <= q) ? g[st][r] * expf(Ac[q] - Ac[s]) : 0.f;
      Esh[q][s] = f2bf(v);
    }
  }
  __syncthreads();

  // ---- Yd = E * Xdt  (64x64, K=NQ) ----
  v8f yd[4] = {};
  for (int k0 = 0; k0 < NQ; k0 += 32){
    v16bf a = frag_a(&Esh[m0][0] + k0, 72, lane);
#pragma unroll
    for (int pt = 0; pt < 4; pt++){
      v16bf b = frag_b(&XdtT[pt * 16][0] + k0, 72, lane);
      yd[pt] = WMMA_BF16(a, b, yd[pt]);
    }
  }
  float* Yrow = Ybuf + ((long)bb * L_ + (long)cc * NQ) * DIN + h * HD;
#pragma unroll
  for (int pt = 0; pt < 4; pt++)
#pragma unroll
    for (int r = 0; r < 8; r++){
      const int q = m0 + 8 * half + r;
      const int p = pt * 16 + nl;
      Yrow[(long)q * DIN + p] = yd[pt][r];
    }

  // ---- states[p,n] = sum_q decay[q]*Xdt[q,p]*Bc[q,n]  (64x128, K=NQ) ----
  v8f sacc[8] = {};
  for (int k0 = 0; k0 < NQ; k0 += 32){
    v16bf a = frag_a(&XdecT[m0][0] + k0, 72, lane);
#pragma unroll
    for (int nt = 0; nt < 8; nt++){
      v16bf b = frag_b(&BcT[nt * 16][0] + k0, 72, lane);
      sacc[nt] = WMMA_BF16(a, b, sacc[nt]);
    }
  }
  float* Sp = states + ((((long)bb * NC + cc) * NH + h) * HD) * DST;
#pragma unroll
  for (int nt = 0; nt < 8; nt++)
#pragma unroll
    for (int r = 0; r < 8; r++){
      const int p = m0 + 8 * half + r;
      const int n = nt * 16 + nl;
      Sp[(long)p * DST + n] = sacc[nt][r];
    }
}

// sequential inter-chunk scan per (b,h): prevs[c] = carry ; carry = dec*carry + st
__global__ void scan_kernel(const float* __restrict__ states,
                            const float* __restrict__ acum,
                            unsigned short* __restrict__ prevs){
  const int h  = blockIdx.x & (NH - 1);
  const int bb = blockIdx.x >> 5;
  const int tid = threadIdx.x;   // 256 threads ; HD*DST = 8192 = 256*32
  float carry[32];
#pragma unroll
  for (int j = 0; j < 32; j++) carry[j] = 0.f;
  for (int cc = 0; cc < NC; cc++){
    const float dec = expf(acum[(((long)bb * NH + h) * NC + cc) * NQ + (NQ - 1)]);
    const long base = (((long)bb * NC + cc) * NH + h) * (HD * DST);
#pragma unroll
    for (int j = 0; j < 32; j++){
      const long e = base + tid + 256 * j;
      const float stv = states[e];
      prevs[e] = f2bf(carry[j]);
      carry[j] = dec * carry[j] + stv;
    }
  }
}

// Yo = exp(Ac[q]) * Cc*prevs^T (WMMA) ; Y += Yo + D_skip * x_ssm
__global__ void __launch_bounds__(128)
chunk_out_kernel(const unsigned short* __restrict__ xbc,
                 const unsigned short* __restrict__ prevs,
                 const float* __restrict__ acum,
                 const float* __restrict__ Dskip,
                 float* __restrict__ Ybuf){
  const int h  = blockIdx.x & (NH - 1);
  const int cc = (blockIdx.x >> 5) & (NC - 1);
  const int bb = blockIdx.x >> 11;
  const int tid = threadIdx.x, lane = tid & 31, wave = tid >> 5;
  const unsigned short* xrow = xbc + ((long)bb * L_ + (long)cc * NQ) * CONVD;
  const unsigned short* Ccm  = xrow + DIN + DST;
  const unsigned short* Xs   = xrow + h * HD;
  const unsigned short* Pv   = prevs + (((long)bb * NC + cc) * NH + h) * (HD * DST);
  __shared__ float Ac[NQ];
  if (tid < NQ) Ac[tid] = acum[(((long)bb * NH + h) * NC + cc) * NQ + tid];
  __syncthreads();
  const int m0 = wave * 16;
  v8f o[4] = {};
  for (int k0 = 0; k0 < DST; k0 += 32){
    v16bf a = frag_a(Ccm + (long)m0 * CONVD + k0, CONVD, lane);
#pragma unroll
    for (int pt = 0; pt < 4; pt++){
      v16bf b = frag_b(Pv + (long)(pt * 16) * DST + k0, DST, lane);
      o[pt] = WMMA_BF16(a, b, o[pt]);
    }
  }
  const int half = lane >> 4, nl = lane & 15;
  const float ds = Dskip[h];
  float* Yrow = Ybuf + ((long)bb * L_ + (long)cc * NQ) * DIN + h * HD;
#pragma unroll
  for (int pt = 0; pt < 4; pt++)
#pragma unroll
    for (int r = 0; r < 8; r++){
      const int q = m0 + 8 * half + r;
      const int p = pt * 16 + nl;
      Yrow[(long)q * DIN + p] += expf(Ac[q]) * o[pt][r] + ds * bf2f(Xs[q * CONVD + p]);
    }
}

// y = Y * silu(z) ; RMSNorm(d_inner) * norm_w -> bf16 for out_proj GEMM
__global__ void gate_norm_kernel(const float* __restrict__ Ybuf,
                                 const float* __restrict__ zx,
                                 const float* __restrict__ norm_w,
                                 unsigned short* __restrict__ ybf){
  const long row = blockIdx.x;
  const int tid = threadIdx.x;
  __shared__ float sbuf[16];
  float vals[8], ss = 0.f;
#pragma unroll
  for (int j = 0; j < 8; j++){
    const int i = tid + 256 * j;
    const float zv = zx[row * DPROJ + i];
    const float yv = Ybuf[row * DIN + i] * (zv / (1.f + expf(-zv)));
    vals[j] = yv;
    ss += yv * yv;
  }
  const float tot = block_reduce(ss, sbuf);
  const float scale = rsqrtf(tot / (float)DIN + 1e-5f);
#pragma unroll
  for (int j = 0; j < 8; j++){
    const int i = tid + 256 * j;
    ybf[row * DIN + i] = f2bf(vals[j] * scale * norm_w[i]);
  }
}

// LayerNorm(D) with ln_w/ln_b + residual x -> d_out
__global__ void final_ln_kernel(const float* __restrict__ outb,
                                const float* __restrict__ x,
                                const float* __restrict__ lnw,
                                const float* __restrict__ lnb,
                                float* __restrict__ dout){
  const long row = blockIdx.x;
  const int tid = threadIdx.x;
  __shared__ float sbuf[16];
  float v[4], s = 0.f;
#pragma unroll
  for (int j = 0; j < 4; j++){ const int d = tid + 256 * j; v[j] = outb[row * D_ + d]; s += v[j]; }
  const float mu = block_reduce(s, sbuf) / (float)D_;
  float s2 = 0.f;
#pragma unroll
  for (int j = 0; j < 4; j++){ const float t = v[j] - mu; s2 += t * t; }
  const float var = block_reduce(s2, sbuf) / (float)D_;
  const float rs = rsqrtf(var + 1e-5f);
#pragma unroll
  for (int j = 0; j < 4; j++){
    const int d = tid + 256 * j;
    dout[row * D_ + d] = x[row * D_ + d] + (v[j] - mu) * rs * lnw[d] + lnb[d];
  }
}

// ---------------- host launcher --------------------------------------------
extern "C" void kernel_launch(void* const* d_in, const int* in_sizes, int n_in,
                              void* d_out, int out_size, void* d_ws, size_t ws_size,
                              hipStream_t stream){
  const float* x      = (const float*)d_in[0];
  const float* winp   = (const float*)d_in[1];
  const float* convw  = (const float*)d_in[2];
  const float* convb  = (const float*)d_in[3];
  const float* dtbias = (const float*)d_in[4];
  const float* Alog   = (const float*)d_in[5];
  const float* Dskip  = (const float*)d_in[6];
  const float* normw  = (const float*)d_in[7];
  const float* woutp  = (const float*)d_in[8];
  const float* lnw    = (const float*)d_in[9];
  const float* lnb    = (const float*)d_in[10];
  float* out = (float*)d_out;

  char* w = (char*)d_ws;
  size_t off = 0;
  auto alloc = [&](size_t bytes) -> char* {
    char* p = w + off; off += (bytes + 255) & ~(size_t)255; return p;
  };
  unsigned short* xbf    = (unsigned short*)alloc((size_t)B_ * L_ * D_ * 2);
  unsigned short* winbf  = (unsigned short*)alloc((size_t)DPROJ * D_ * 2);
  unsigned short* woutbf = (unsigned short*)alloc((size_t)D_ * DIN * 2);
  float*          zx     = (float*)         alloc((size_t)B_ * L_ * DPROJ * 4);
  unsigned short* xbc    = (unsigned short*)alloc((size_t)B_ * L_ * CONVD * 2);
  float*          dtb    = (float*)         alloc((size_t)B_ * L_ * NH * 4);
  float*          acum   = (float*)         alloc((size_t)B_ * NH * NC * NQ * 4);
  float*          states = (float*)         alloc((size_t)B_ * NC * NH * HD * DST * 4);
  unsigned short* prevs  = (unsigned short*)alloc((size_t)B_ * NC * NH * HD * DST * 2);
  float*          Ybuf   = (float*)         alloc((size_t)B_ * L_ * DIN * 4);
  unsigned short* ybf    = (unsigned short*)alloc((size_t)B_ * L_ * DIN * 2);
  float*          outb   = (float*)         alloc((size_t)B_ * L_ * D_ * 4);
  (void)in_sizes; (void)n_in; (void)out_size; (void)ws_size;

  { long n = (long)B_ * L_ * D_;
    cvt_bf16_kernel<<<(unsigned)((n + 255) / 256), 256, 0, stream>>>(x, xbf, n); }
  { long n = (long)DPROJ * D_;
    cvt_bf16_kernel<<<(unsigned)((n + 255) / 256), 256, 0, stream>>>(winp, winbf, n); }
  { long n = (long)D_ * DIN;
    cvt_bf16_kernel<<<(unsigned)((n + 255) / 256), 256, 0, stream>>>(woutp, woutbf, n); }

  { const int M = B_ * L_, N = DPROJ, K = D_;
    const int blocks = (M / 128) * (N / 32);
    gemm_bf16_wmma<<<blocks, 128, 0, stream>>>(xbf, winbf, zx, M, N, K, K, K, N); }

  { long n = (long)B_ * L_ * CONVD;
    conv_silu_kernel<<<(unsigned)((n + 255) / 256), 256, 0, stream>>>(zx, convw, convb, xbc); }

  dt_acum_kernel<<<B_ * NH * NC, NQ, 0, stream>>>(zx, dtbias, Alog, dtb, acum);

  chunk_fwd_kernel<<<B_ * NC * NH, 128, 0, stream>>>(xbc, dtb, acum, Ybuf, states);

  scan_kernel<<<B_ * NH, 256, 0, stream>>>(states, acum, prevs);

  chunk_out_kernel<<<B_ * NC * NH, 128, 0, stream>>>(xbc, prevs, acum, Dskip, Ybuf);

  gate_norm_kernel<<<B_ * L_, 256, 0, stream>>>(Ybuf, zx, normw, ybf);

  { const int M = B_ * L_, N = D_, K = DIN;
    const int blocks = (M / 128) * (N / 32);
    gemm_bf16_wmma<<<blocks, 128, 0, stream>>>(ybf, woutbf, outb, M, N, K, K, K, N); }

  final_ln_kernel<<<B_ * L_, 256, 0, stream>>>(outb, x, lnw, lnb, out);
}
